// SS1D_16492674417413
// MI455X (gfx1250) — compile-verified, matrix operand order checked by reading, not run on hardware
//
#include <hip/hip_runtime.h>
#include <hip/hip_bf16.h>

// ---------------- problem constants (match reference) ----------------
#define B_SZ   2
#define T_SZ   2048
#define DM     256          // D_MODEL
#define DI     512          // D_INNER
#define NST    16           // D_STATE
#define RANK   16           // DT_RANK
#define NC     48           // RANK + 2*NST rows of x_dbl
#define KDIR   2
#define NSEG   16           // scan segments
#define SEGLEN (T_SZ / NSEG)  // 128 steps per segment
#define SCHUNK 32           // LDS staging chunk (timesteps)

typedef __attribute__((ext_vector_type(16))) _Float16 v16h;
typedef __attribute__((ext_vector_type(8)))  _Float16 v8h;
typedef __attribute__((ext_vector_type(8)))  float    v8f;

// ---------------------------------------------------------------------
// f32 -> f16 conversion
// ---------------------------------------------------------------------
__global__ void ss1d_f32_to_f16(const float* __restrict__ in,
                                _Float16* __restrict__ out, int n) {
    int i = blockIdx.x * blockDim.x + threadIdx.x;
    if (i < n) out[i] = (_Float16)in[i];
}

// ---------------------------------------------------------------------
// WMMA fragment loader: 16(major) x 32(K) f16 fragment, wave32.
// ISA layout (16-bit A 16x32 / B 32x16): lane = major%16, half = lane/16;
//   elems 0..7  -> K = half*8 .. half*8+7
//   elems 8..15 -> K = 16+half*8 .. 16+half*8+7
// K-contiguous in memory for (major x K) row-major -> two b128 loads.
// ---------------------------------------------------------------------
__device__ __forceinline__ v16h ss1d_load_frag(const _Float16* __restrict__ base,
                                               int ld, int major0, int k0, int lane) {
    int mj   = lane & 15;
    int half = lane >> 4;
    const _Float16* p = base + (long)(major0 + mj) * ld + k0;
    v8h lo = *(const v8h*)(p + half * 8);
    v8h hi = *(const v8h*)(p + 16 + half * 8);
    return __builtin_shufflevector(lo, hi, 0,1,2,3,4,5,6,7,8,9,10,11,12,13,14,15);
}

// ---------------------------------------------------------------------
// Batched WMMA GEMM:  C[m,n] = sum_k A[m,k] * B[n,k]   (B given N x K)
// Block = 128 threads = 4 waves; each wave computes a 16 x 64 strip.
// All B fragments for a k-step are preloaded before the WMMAs so the
// scheduler can overlap WMMA j with loads j+1..3 (partial loadcnt waits).
// Requires K % 32 == 0, N % 256 == 0, M % 16 == 0.
// ---------------------------------------------------------------------
__global__ void ss1d_wmma_gemm_bt(const _Float16* __restrict__ A,
                                  const _Float16* __restrict__ B,
                                  float* __restrict__ C,
                                  int M, int N, int K,
                                  int lda, int ldb, int ldc,
                                  int abatch, int strideA, int strideB, int strideC) {
    int bz = blockIdx.z;
    A += (long)(bz % abatch) * strideA;
    B += (long)bz * strideB;
    C += (long)bz * strideC;

    int wave = threadIdx.x >> 5;
    int lane = threadIdx.x & 31;
    int row0    = blockIdx.y * 16;
    int colBase = (blockIdx.x * 4 + wave) * 64;
    if (row0 >= M || colBase >= N) return;   // uniform per wave

    v8f acc[4] = {};
    for (int k0 = 0; k0 < K; k0 += 32) {
        v16h a = ss1d_load_frag(A, lda, row0, k0, lane);
        v16h bf[4];
#pragma unroll
        for (int j = 0; j < 4; ++j)
            bf[j] = ss1d_load_frag(B, ldb, colBase + j * 16, k0, lane);
        if (k0 + 32 < K) {   // pull next A k-tile line toward the WGP cache
            __builtin_prefetch(A + (long)(row0 + (lane & 15)) * lda + k0 + 32, 0, 3);
        }
#pragma unroll
        for (int j = 0; j < 4; ++j)
            acc[j] = __builtin_amdgcn_wmma_f32_16x16x32_f16(
                false, a, false, bf[j], (short)0, acc[j], false, false);
    }
    // C/D layout: VGPR v -> M = v (lanes 0-15) or 8+v (lanes 16-31); N = lane%16
    int n  = lane & 15;
    int mb = (lane >> 4) * 8;
#pragma unroll
    for (int j = 0; j < 4; ++j) {
        float* cp = C + (long)(row0 + mb) * ldc + colBase + j * 16 + n;
#pragma unroll
        for (int v = 0; v < 8; ++v) cp[(long)v * ldc] = acc[j][v];
    }
}

// ---------------------------------------------------------------------
// Depthwise conv1d (3-tap, pad 1) + SiLU on xp = xz[..., :DI].
// Writes xc (b,t,d) f32 and f16 copies for both scan directions in
// (bz, l, d) layout (d contiguous => same BT fragment path for GEMM-B).
// ---------------------------------------------------------------------
__global__ void ss1d_conv_silu(const float* __restrict__ xz,
                               const float* __restrict__ cw,
                               const float* __restrict__ cb,
                               float* __restrict__ xc,
                               _Float16* __restrict__ xs_h) {
    int idx = blockIdx.x * blockDim.x + threadIdx.x;
    if (idx >= B_SZ * T_SZ * DI) return;
    int d = idx % DI;
    int t = (idx / DI) % T_SZ;
    int b = idx / (DI * T_SZ);

    const float* xp = xz + (long)b * T_SZ * (2 * DI);
    float acc = cb[d];
#pragma unroll
    for (int j = 0; j < 3; ++j) {
        int tt = t + j - 1;
        if (tt >= 0 && tt < T_SZ) acc += cw[d * 3 + j] * xp[(long)tt * (2 * DI) + d];
    }
    float v = acc / (1.f + __expf(-acc));       // SiLU
    xc[((long)b * T_SZ + t) * DI + d] = v;
    _Float16 hv = (_Float16)v;
    xs_h[(((long)(b * 2 + 0)) * T_SZ + t) * DI + d]              = hv;  // fwd
    xs_h[(((long)(b * 2 + 1)) * T_SZ + (T_SZ - 1 - t)) * DI + d] = hv;  // rev
}

// ---------------------------------------------------------------------
// Segmented selective scan.  h_l = exp(delta_l*A)*h_{l-1} + delta_l*u_l*B_l
// is linear with a DIAGONAL transfer, and prod_l exp(delta_l*A_n) =
// exp(A_n * sum_l delta_l): a segment summary is just q[16] + sum_delta.
//
// Pass 1: per (b,dir,seg) block, run segment from h=0 -> q, sum_delta.
// Pass 2: per (b,dir) block, compose 16 segment summaries -> h_init[seg].
// Pass 3: per (b,dir,seg) block, replay from exact h_init, emit y.
// Serial path: 2048 -> ~272 steps; parallelism: 4 -> 64 blocks.
// ---------------------------------------------------------------------
__device__ __forceinline__ float ss1d_softplus(float x) {
    return (x > 20.f) ? x : log1pf(__expf(x));
}

__global__ void ss1d_scan_part1(const float* __restrict__ xdbl,
                                const float* __restrict__ xc,
                                const float* __restrict__ dtw,
                                const float* __restrict__ dtb,
                                const float* __restrict__ A_logs,
                                float* __restrict__ qstate,    // (bz,seg,d,16)
                                float* __restrict__ sumdelta)  // (bz,seg,d)
{
    int blk = blockIdx.x;                 // bz*NSEG + seg
    int bz = blk / NSEG, seg = blk % NSEG;
    int b = bz >> 1, k = bz & 1;
    int d = threadIdx.x;                  // 0..511

    float w[RANK], Ac[NST], h[NST];
#pragma unroll
    for (int r = 0; r < RANK; ++r) w[r] = dtw[((long)k * DI + d) * RANK + r];
#pragma unroll
    for (int n = 0; n < NST; ++n) {
        Ac[n] = -__expf(A_logs[((long)k * DI + d) * NST + n]);
        h[n] = 0.f;
    }
    float bias = dtb[k * DI + d];
    float sd = 0.f;

    __shared__ float sl[(RANK + NST) * SCHUNK];   // dt rows + B rows only: 4 KB
    const float* xd = xdbl + (long)bz * NC * T_SZ;

    for (int l0 = seg * SEGLEN; l0 < (seg + 1) * SEGLEN; l0 += SCHUNK) {
        __syncthreads();
        for (int i = threadIdx.x; i < (RANK + NST) * SCHUNK; i += DI) {
            int c = i / SCHUNK, lc = i % SCHUNK;
            sl[c * SCHUNK + lc] = xd[(long)c * T_SZ + l0 + lc];
        }
        __syncthreads();
        for (int lc = 0; lc < SCHUNK; ++lc) {
            int l = l0 + lc;
            int t = k ? (T_SZ - 1 - l) : l;
            float u = xc[((long)b * T_SZ + t) * DI + d];
            float dtr = bias;
#pragma unroll
            for (int r = 0; r < RANK; ++r) dtr += w[r] * sl[r * SCHUNK + lc];
            float delta = ss1d_softplus(dtr);
            sd += delta;
            float du = delta * u;
#pragma unroll
            for (int n = 0; n < NST; ++n) {
                h[n] = __expf(delta * Ac[n]) * h[n] + du * sl[(RANK + n) * SCHUNK + lc];
            }
        }
    }
#pragma unroll
    for (int n = 0; n < NST; ++n) qstate[((long)blk * DI + d) * NST + n] = h[n];
    sumdelta[(long)blk * DI + d] = sd;
}

__global__ void ss1d_scan_part2(const float* __restrict__ A_logs,
                                const float* __restrict__ qstate,
                                const float* __restrict__ sumdelta,
                                float* __restrict__ hinit)     // (bz,seg,d,16)
{
    int bz = blockIdx.x;                  // 4 blocks
    int k = bz & 1;
    int d = threadIdx.x;
    float Ac[NST], h[NST];
#pragma unroll
    for (int n = 0; n < NST; ++n) {
        Ac[n] = -__expf(A_logs[((long)k * DI + d) * NST + n]);
        h[n] = 0.f;
    }
    for (int s = 0; s < NSEG; ++s) {
        long base = ((long)(bz * NSEG + s) * DI + d) * NST;
#pragma unroll
        for (int n = 0; n < NST; ++n) hinit[base + n] = h[n];
        float sd = sumdelta[(long)(bz * NSEG + s) * DI + d];
#pragma unroll
        for (int n = 0; n < NST; ++n) {
            h[n] = __expf(Ac[n] * sd) * h[n] + qstate[base + n];
        }
    }
}

__global__ void ss1d_scan_part3(const float* __restrict__ xdbl,
                                const float* __restrict__ xc,
                                const float* __restrict__ dtw,
                                const float* __restrict__ dtb,
                                const float* __restrict__ A_logs,
                                const float* __restrict__ Ds,
                                const float* __restrict__ hinit,
                                float* __restrict__ yk)        // (bz,t,d)
{
    int blk = blockIdx.x;                 // bz*NSEG + seg
    int bz = blk / NSEG, seg = blk % NSEG;
    int b = bz >> 1, k = bz & 1;
    int d = threadIdx.x;

    float w[RANK], Ac[NST], h[NST];
#pragma unroll
    for (int r = 0; r < RANK; ++r) w[r] = dtw[((long)k * DI + d) * RANK + r];
#pragma unroll
    for (int n = 0; n < NST; ++n) {
        Ac[n] = -__expf(A_logs[((long)k * DI + d) * NST + n]);
        h[n] = hinit[((long)blk * DI + d) * NST + n];
    }
    float bias = dtb[k * DI + d];
    float Dd   = Ds[k * DI + d];

    __shared__ float sl[NC * SCHUNK];     // dt + B + C rows: 6 KB
    const float* xd = xdbl + (long)bz * NC * T_SZ;

    for (int l0 = seg * SEGLEN; l0 < (seg + 1) * SEGLEN; l0 += SCHUNK) {
        __syncthreads();
        for (int i = threadIdx.x; i < NC * SCHUNK; i += DI) {
            int c = i / SCHUNK, lc = i % SCHUNK;
            sl[c * SCHUNK + lc] = xd[(long)c * T_SZ + l0 + lc];
        }
        __syncthreads();
        for (int lc = 0; lc < SCHUNK; ++lc) {
            int l = l0 + lc;
            int t = k ? (T_SZ - 1 - l) : l;
            float u = xc[((long)b * T_SZ + t) * DI + d];
            float dtr = bias;
#pragma unroll
            for (int r = 0; r < RANK; ++r) dtr += w[r] * sl[r * SCHUNK + lc];
            float delta = ss1d_softplus(dtr);
            float du = delta * u;
            float acc = 0.f;
#pragma unroll
            for (int n = 0; n < NST; ++n) {
                h[n] = __expf(delta * Ac[n]) * h[n] + du * sl[(RANK + n) * SCHUNK + lc];
                acc += h[n] * sl[(RANK + NST + n) * SCHUNK + lc];
            }
            yk[((long)bz * T_SZ + t) * DI + d] = acc + Dd * u;
        }
    }
}

// ---------------------------------------------------------------------
// Merge directions + LayerNorm(Di) + SiLU(z) gate; emit f16 for out_proj.
// ---------------------------------------------------------------------
__global__ void ss1d_ln_gate(const float* __restrict__ yk,
                             const float* __restrict__ xz,
                             const float* __restrict__ g,
                             const float* __restrict__ be,
                             _Float16* __restrict__ yg) {
    int row = blockIdx.x;                 // b*T + t
    int b = row / T_SZ, t = row % T_SZ;
    int i = threadIdx.x;                  // 0..255

    const float* y0 = yk + (((long)(b * 2 + 0)) * T_SZ + t) * DI;
    const float* y1 = yk + (((long)(b * 2 + 1)) * T_SZ + t) * DI;

    float v0 = y0[i]       + y1[i];
    float v1 = y0[i + 256] + y1[i + 256];

    __shared__ float ssum[256], ssq[256];
    ssum[i] = v0 + v1;
    ssq[i]  = v0 * v0 + v1 * v1;
    __syncthreads();
    for (int off = 128; off > 0; off >>= 1) {
        if (i < off) { ssum[i] += ssum[i + off]; ssq[i] += ssq[i + off]; }
        __syncthreads();
    }
    float mu   = ssum[0] * (1.f / DI);
    float var  = ssq[0] * (1.f / DI) - mu * mu;
    float rstd = rsqrtf(var + 1e-5f);

    const float* zrow = xz + (long)row * (2 * DI) + DI;
#pragma unroll
    for (int s = 0; s < 2; ++s) {
        int dd = i + s * 256;
        float v  = (s == 0) ? v0 : v1;
        float yn = (v - mu) * rstd * g[dd] + be[dd];
        float z  = zrow[dd];
        float gate = z / (1.f + __expf(-z));
        yg[(long)row * DI + dd] = (_Float16)(yn * gate);
    }
}

// ---------------------------------------------------------------------
extern "C" void kernel_launch(void* const* d_in, const int* in_sizes, int n_in,
                              void* d_out, int out_size, void* d_ws, size_t ws_size,
                              hipStream_t stream) {
    const float* x       = (const float*)d_in[0];   // (B,T,256)
    const float* W_in    = (const float*)d_in[1];   // (1024,256)
    const float* conv_w  = (const float*)d_in[2];   // (512,1,3)
    const float* conv_b  = (const float*)d_in[3];   // (512)
    const float* xprojw  = (const float*)d_in[4];   // (2,48,512)
    const float* dtw     = (const float*)d_in[5];   // (2,512,16)
    const float* dtb     = (const float*)d_in[6];   // (2,512)
    const float* A_logs  = (const float*)d_in[7];   // (2,512,16)
    const float* Ds      = (const float*)d_in[8];   // (2,512)
    const float* ln_g    = (const float*)d_in[9];   // (512)
    const float* ln_b    = (const float*)d_in[10];  // (512)
    const float* W_out   = (const float*)d_in[11];  // (256,512)
    float* out = (float*)d_out;

    const int MROWS = B_SZ * T_SZ;                  // 4096 token rows

    // ---- carve workspace (256B aligned slabs) ----
    char* p = (char*)d_ws;
    auto carve = [&](size_t bytes) { void* r = (void*)p; p += (bytes + 255) & ~(size_t)255; return r; };
    _Float16* xh      = (_Float16*)carve((size_t)MROWS * DM * 2);
    _Float16* Win_h   = (_Float16*)carve((size_t)2 * DI * DM * 2);
    _Float16* xpw_h   = (_Float16*)carve((size_t)KDIR * NC * DI * 2);
    _Float16* Wout_h  = (_Float16*)carve((size_t)DM * DI * 2);
    float*    xz      = (float*)   carve((size_t)MROWS * 2 * DI * 4);
    float*    xc      = (float*)   carve((size_t)B_SZ * T_SZ * DI * 4);
    _Float16* xs_h    = (_Float16*)carve((size_t)B_SZ * KDIR * T_SZ * DI * 2);
    float*    xdbl    = (float*)   carve((size_t)B_SZ * KDIR * NC * T_SZ * 4);
    float*    yk      = (float*)   carve((size_t)B_SZ * KDIR * T_SZ * DI * 4);
    _Float16* yg      = (_Float16*)carve((size_t)MROWS * DI * 2);
    float*    qstate  = (float*)   carve((size_t)B_SZ * KDIR * NSEG * DI * NST * 4);
    float*    sdelta  = (float*)   carve((size_t)B_SZ * KDIR * NSEG * DI * 4);
    float*    hinit   = (float*)   carve((size_t)B_SZ * KDIR * NSEG * DI * NST * 4);

    // ---- 1. f16 conversions ----
    auto cvt = [&](const float* src, _Float16* dst, int n) {
        ss1d_f32_to_f16<<<(n + 255) / 256, 256, 0, stream>>>(src, dst, n);
    };
    cvt(x,      xh,     MROWS * DM);
    cvt(W_in,   Win_h,  2 * DI * DM);
    cvt(xprojw, xpw_h,  KDIR * NC * DI);
    cvt(W_out,  Wout_h, DM * DI);

    // ---- 2. in_proj: xz(4096x1024) = xh(4096x256) @ W_in^T ----
    ss1d_wmma_gemm_bt<<<dim3((2 * DI) / 256, MROWS / 16, 1), 128, 0, stream>>>(
        xh, Win_h, xz, MROWS, 2 * DI, DM, DM, DM, 2 * DI, 1, 0, 0, 0);

    // ---- 3. depthwise conv + SiLU + f16 bidirectional activations ----
    ss1d_conv_silu<<<(B_SZ * T_SZ * DI + 255) / 256, 256, 0, stream>>>(
        xz, conv_w, conv_b, xc, xs_h);

    // ---- 4. x_proj: per (b,dir): xdbl(48x2048) = xpw(48x512) @ xs ----
    ss1d_wmma_gemm_bt<<<dim3(T_SZ / 256, NC / 16, B_SZ * KDIR), 128, 0, stream>>>(
        xpw_h, xs_h, xdbl, NC, T_SZ, DI, DI, DI, T_SZ,
        KDIR, NC * DI, T_SZ * DI, NC * T_SZ);

    // ---- 5. segmented selective scan ----
    ss1d_scan_part1<<<B_SZ * KDIR * NSEG, DI, 0, stream>>>(
        xdbl, xc, dtw, dtb, A_logs, qstate, sdelta);
    ss1d_scan_part2<<<B_SZ * KDIR, DI, 0, stream>>>(
        A_logs, qstate, sdelta, hinit);
    ss1d_scan_part3<<<B_SZ * KDIR * NSEG, DI, 0, stream>>>(
        xdbl, xc, dtw, dtb, A_logs, Ds, hinit, yk);

    // ---- 6. merge + LayerNorm + SiLU gate -> f16 ----
    ss1d_ln_gate<<<MROWS, 256, 0, stream>>>(yk, xz, ln_g, ln_b, yg);

    // ---- 7. out_proj: out(4096x256) = yg(4096x512) @ W_out^T ----
    ss1d_wmma_gemm_bt<<<dim3(DM / 256, MROWS / 16, 1), 128, 0, stream>>>(
        yg, Wout_h, out, MROWS, DM, DI, DI, DI, DM, 1, 0, 0, 0);
}
